// Defaultloss_27925877358880
// MI455X (gfx1250) — compile-verified, matrix operand order checked by reading, not run on hardware
//
#include <hip/hip_runtime.h>
#include <math.h>

// ---------------- problem constants (match reference) ----------------
#define BB      8
#define AN      100000          // anchors
#define MM      100             // gt boxes
#define NCLSV   80
#define CHN     85              // 5 + NCLS
#define NCHCLS  81              // 1 + NCLS
#define EPSV    1e-4f

// log constants:  L0 = log(1-e)-log(e), C0 = -log(1-e), C1 = -log(e)
#define LOG_EPS_F    (-9.210340371976182f)
#define LOG_1ME_F    (-1.0000500033334732e-4f)
#define L0C          (LOG_1ME_F - LOG_EPS_F)     // 9.21024037
#define C0C          (-LOG_1ME_F)
#define C1C          (-LOG_EPS_F)

#define ANCH_PER_WAVE   16
#define ANCH_PER_BLOCK  128     // 8 waves * 16 anchors
#define BLOCKS_PER_B    ((AN + ANCH_PER_BLOCK - 1) / ANCH_PER_BLOCK)  // 782

typedef __attribute__((ext_vector_type(16))) _Float16 v16h;
typedef __attribute__((ext_vector_type(8)))  float    v8f;

__device__ __forceinline__ float clampe(float x) {
    return fminf(fmaxf(x, EPSV), 1.0f - EPSV);
}

// ---------------------------------------------------------------------
// Kernel 1: per-(b, anchor-tile) partial sums.
// Channel-sum over cls channels done on the matrix pipe:
//   D = ones(16x32) x B(32ch x 16anchors)  ->  C[0][lane] = per-anchor sum.
// fp16 rounding residual is accumulated in fp32 and added back -> fp32 accuracy.
// ---------------------------------------------------------------------
__global__ __launch_bounds__(256)
void loss_partial_kernel(const float* __restrict__ dt,
                         const float* __restrict__ gt,
                         const float* __restrict__ anchors,
                         const int*   __restrict__ assign,
                         float*       __restrict__ partials)
{
    const int blk  = blockIdx.x % BLOCKS_PER_B;
    const int b    = blockIdx.x / BLOCKS_PER_B;
    const int tid  = threadIdx.x;
    const int wave = tid >> 5;
    const int lane = tid & 31;
    const int n    = lane & 15;     // anchor slot within wave (N of WMMA)
    const int half = lane >> 4;     // which K-half this lane feeds

    const int  a_raw = blk * ANCH_PER_BLOCK + wave * ANCH_PER_WAVE + n;
    const bool a_ok  = (a_raw < AN);
    const int  aa    = a_ok ? a_raw : (AN - 1);   // clamp: safe address, masked later

    const size_t dtB = (size_t)b * CHN * AN;

    // ---- WMMA: sum of clamped cls channels k = 0..63 (dt channels 4..67) ----
    v16h ones;
    #pragma unroll
    for (int j = 0; j < 16; ++j) ones[j] = (_Float16)1.0f;

    v8f   acc  = {};
    float corr = 0.0f;   // fp16 rounding residual (fp32)

    #pragma unroll
    for (int k0 = 0; k0 < 64; k0 += 32) {
        const float* pB = dt + dtB + (size_t)(4 + k0 + half * 16) * AN + aa;
        v16h bv;
        #pragma unroll
        for (int j = 0; j < 16; ++j) {
            float x  = __builtin_nontemporal_load(pB + (size_t)j * AN);
            float xc = clampe(x);
            _Float16 h = (_Float16)xc;
            bv[j] = h;
            corr += xc - (float)h;
        }
        // (neg_a, A, neg_b, B, c_mod, C, reuse_a, reuse_b)
        acc = __builtin_amdgcn_wmma_f32_16x16x32_f16(false, ones, false, bv,
                                                     (short)0, acc, false, false);
    }

    // ---- tail: cls channels k = 64..80 (dt channels 68..84), plain fp32 ----
    float tail = 0.0f;
    #pragma unroll
    for (int k = 64; k < NCHCLS; ++k) {
        float x = __builtin_nontemporal_load(dt + dtB + (size_t)(4 + k) * AN + aa);
        tail += clampe(x);
    }

    // partner half-lane holds this anchor's other 32 residuals
    corr += __shfl_xor(corr, 16, 32);
    const float s_cls = acc[0] + corr + tail;   // fp32-accurate 81-channel sum

    // ------------------------- per-anchor epilogue -------------------------
    int afi = assign[(size_t)b * AN + aa];
    if (!a_ok) afi = -1;
    const bool pc = (afi >= 0);   // cls-counted (assigned or background)
    const bool pb = (afi >= 1);   // box-counted (assigned to a gt)

    int idx = afi - 1;
    idx = idx < 0 ? 0 : (idx > MM - 1 ? MM - 1 : idx);
    const float* g = gt + ((size_t)b * MM + idx) * 5;
    const float gx = g[0], gy = g[1], gw = g[2], gh = g[3];
    int cid = (int)g[4];                         // 1..NCLS
    int chc = 4 + cid;
    chc = chc < 5 ? 5 : (chc > CHN - 1 ? CHN - 1 : chc);

    const float pObj = clampe(dt[dtB + (size_t)4   * AN + aa]);
    const float pCls = clampe(dt[dtB + (size_t)chc * AN + aa]);

    // base: all 81 targets at eps; positives flip obj + one class channel
    float cls_per = L0C * s_cls + (float)NCHCLS * C0C;
    cls_per += pb ? (-2.0f * L0C * (pObj + pCls) + 2.0f * (C1C - C0C)) : 0.0f;
    float cls_c  = pc ? cls_per : 0.0f;
    float ncls_c = pc ? 1.0f : 0.0f;
    float nbox_c = pb ? 1.0f : 0.0f;

    // box regression (L2 on encoded offsets)
    const float4 an = ((const float4*)anchors)[aa];
    const float aw = an.z - an.x, ah = an.w - an.y;
    const float ax = an.x + 0.5f * aw, ay = an.y + 0.5f * ah;
    const float tx = (gx + 0.5f * gw - ax) / aw;
    const float ty = (gy + 0.5f * gh - ay) / ah;
    const float tw = logf(gw / aw);
    const float th = logf(gh / ah);
    const float d0 = dt[dtB + (size_t)0 * AN + aa];
    const float d1 = dt[dtB + (size_t)1 * AN + aa];
    const float d2 = dt[dtB + (size_t)2 * AN + aa];
    const float d3 = dt[dtB + (size_t)3 * AN + aa];
    const float e0 = tx - d0, e1 = ty - d1, e2 = tw - d2, e3 = th - d3;
    float box_c = pb ? (e0 * e0 + e1 * e1 + e2 * e2 + e3 * e3) : 0.0f;

    // lanes 16..31 duplicate lanes 0..15 -> zero the mirror before reducing
    if (half) { cls_c = 0.0f; box_c = 0.0f; ncls_c = 0.0f; nbox_c = 0.0f; }
    #pragma unroll
    for (int off = 8; off; off >>= 1) {
        cls_c  += __shfl_down(cls_c,  off, 32);
        box_c  += __shfl_down(box_c,  off, 32);
        ncls_c += __shfl_down(ncls_c, off, 32);
        nbox_c += __shfl_down(nbox_c, off, 32);
    }

    __shared__ float sm[8][4];
    if (lane == 0) {
        sm[wave][0] = cls_c;  sm[wave][1] = box_c;
        sm[wave][2] = ncls_c; sm[wave][3] = nbox_c;
    }
    __syncthreads();
    if (tid == 0) {
        float c = 0.0f, bx = 0.0f, nc = 0.0f, nb = 0.0f;
        #pragma unroll
        for (int w = 0; w < 8; ++w) {
            c += sm[w][0]; bx += sm[w][1]; nc += sm[w][2]; nb += sm[w][3];
        }
        float* p = partials + (size_t)blockIdx.x * 4;
        p[0] = c; p[1] = bx; p[2] = nc; p[3] = nb;
    }
}

// ---------------------------------------------------------------------
// Kernel 2: deterministic final reduction (single block, fixed tree).
// ---------------------------------------------------------------------
__global__ __launch_bounds__(256)
void loss_final_kernel(const float* __restrict__ partials,
                       float*       __restrict__ out)
{
    __shared__ float4 sm[256];
    const int tid = threadIdx.x;
    float total = 0.0f;

    for (int b = 0; b < BB; ++b) {
        float4 acc = make_float4(0.0f, 0.0f, 0.0f, 0.0f);
        for (int i = tid; i < BLOCKS_PER_B; i += 256) {
            const float* p = partials + ((size_t)b * BLOCKS_PER_B + i) * 4;
            acc.x += p[0]; acc.y += p[1]; acc.z += p[2]; acc.w += p[3];
        }
        sm[tid] = acc;
        __syncthreads();
        for (int off = 128; off; off >>= 1) {
            if (tid < off) {
                sm[tid].x += sm[tid + off].x;
                sm[tid].y += sm[tid + off].y;
                sm[tid].z += sm[tid + off].z;
                sm[tid].w += sm[tid + off].w;
            }
            __syncthreads();
        }
        if (tid == 0) {
            const float cls_sum = sm[0].x, box_sum = sm[0].y;
            const float ncls = sm[0].z,   nbox = sm[0].w;
            const float cls_loss = cls_sum / fmaxf(ncls, 1.0f);
            const float box_loss = (nbox > 0.0f) ? box_sum / fmaxf(nbox, 1.0f) : 0.0f;
            total += cls_loss + box_loss;
        }
        __syncthreads();
    }
    if (tid == 0) out[0] = total / (float)BB;
}

// ---------------------------------------------------------------------
extern "C" void kernel_launch(void* const* d_in, const int* in_sizes, int n_in,
                              void* d_out, int out_size, void* d_ws, size_t ws_size,
                              hipStream_t stream)
{
    const float* dt      = (const float*)d_in[0];   // [B, 85, A] fp32
    const float* gt      = (const float*)d_in[1];   // [B, M, 5] fp32
    const float* anchors = (const float*)d_in[2];   // [A, 4] fp32
    const int*   assign  = (const int*)  d_in[3];   // [B, A] int32
    float*       out     = (float*)d_out;
    float*       partials = (float*)d_ws;           // [B*BLOCKS_PER_B][4] ~ 100 KB

    (void)in_sizes; (void)n_in; (void)out_size; (void)ws_size;

    loss_partial_kernel<<<dim3(BB * BLOCKS_PER_B), dim3(256), 0, stream>>>(
        dt, gt, anchors, assign, partials);
    loss_final_kernel<<<dim3(1), dim3(256), 0, stream>>>(partials, out);
}
